// mnistautoencoder_60078002536976
// MI455X (gfx1250) — matrix operand order repaired and checked by reading, compiled, not run on hardware
//
#include <hip/hip_runtime.h>
#include <cstdint>
#include <cstddef>

typedef __attribute__((ext_vector_type(16))) __bf16 v16bf;
typedef __attribute__((ext_vector_type(8)))  __bf16 v8bf;
typedef __attribute__((ext_vector_type(8)))  float  v8f;

#define WMMA_BF16(ACC, AF, BF)                                                \
    __builtin_amdgcn_wmma_f32_16x16x32_bf16(false, (AF), false, (BF),         \
                                            (short)0, (ACC), false, false)

// ---------------------------------------------------------------------------
// Pack fp32 activations [B,N] -> zero-padded bf16 [B,Npad]
// ---------------------------------------------------------------------------
__global__ void k_pack_act(const float* __restrict__ X, __bf16* __restrict__ A,
                           int Bn, int N, int Npad) {
    int idx = blockIdx.x * blockDim.x + threadIdx.x;
    if (idx >= Bn * Npad) return;
    int b = idx / Npad, j = idx - b * Npad;
    float v = (j < N) ? X[(size_t)b * N + j] : 0.0f;
    A[idx] = (__bf16)v;
}

// ---------------------------------------------------------------------------
// Pack fp32 weights W[K,N] -> transposed zero-padded bf16 Wt[Npad,Kpad]
// (Wt[n,k] = W[k,n]) so B-fragment loads are contiguous along K.
// ---------------------------------------------------------------------------
__global__ void k_pack_wt(const float* __restrict__ W, __bf16* __restrict__ Wt,
                          int K, int N, int Kpad, int Npad) {
    int idx = blockIdx.x * blockDim.x + threadIdx.x;
    if (idx >= Npad * Kpad) return;
    int n = idx / Kpad, k = idx - n * Kpad;
    float v = (n < N && k < K) ? W[(size_t)k * N + n] : 0.0f;
    Wt[idx] = (__bf16)v;
}

__global__ void k_zero32(uint32_t* __restrict__ p, int n) {
    int i = blockIdx.x * blockDim.x + threadIdx.x;
    if (i < n) p[i] = 0u;
}

// colsq[j] = -0.5 * sum_k W[k,j]^2 (fp32 weights), zero in padding
__global__ void k_colsq(const float* __restrict__ W, int K, int N, int Npad,
                        float* __restrict__ colsq) {
    int j = blockIdx.x * blockDim.x + threadIdx.x;
    if (j >= Npad) return;
    float s = 0.0f;
    if (j < N) {
        for (int k = 0; k < K; ++k) {
            float v = W[(size_t)k * N + j];
            s += v * v;
        }
    }
    colsq[j] = (j < N) ? -0.5f * s : 0.0f;
}

// rowsq[b] = -0.5 * sum_k H[b,k]^2 over padded bf16 row (pads are zero)
__global__ void k_rowsq(const __bf16* __restrict__ H, int Kpad,
                        float* __restrict__ rowsq, int Bn) {
    int b = blockIdx.x * blockDim.x + threadIdx.x;
    if (b >= Bn) return;
    const __bf16* row = H + (size_t)b * Kpad;
    float s = 0.0f;
    for (int k = 0; k < Kpad; ++k) {
        float x = (float)row[k];
        s += x * x;
    }
    rowsq[b] = -0.5f * s;
}

// ---------------------------------------------------------------------------
// Register-blocked WMMA GEMM: one wave computes a 64(M) x 16(N) strip:
//   Y[m,n] = sum_k A[m,k]*Bt[n,k] + rowsq[m] + colsq[n]
// A: bf16 [M, Kpad] row-major; Bt: bf16 [Npad, Kpad] row-major (W transposed).
// Kpad multiple of 32. blockDim = 32 (one full wave; EXEC all ones).
// B fragment loaded once per K-step, reused by 4 independent WMMAs; K-loop is
// manually software-pipelined so next-step loads overlap the WMMA group.
// ---------------------------------------------------------------------------
__global__ __launch_bounds__(32)
void k_gemm_wmma(const __bf16* __restrict__ A, const __bf16* __restrict__ Bt,
                 const float* __restrict__ rowsq, const float* __restrict__ colsq,
                 float* __restrict__ Y, int Kpad, int ldY) {
    const int lane = threadIdx.x & 31;
    const int hf   = lane >> 4;   // half-wave selector
    const int l16  = lane & 15;
    const int m0 = blockIdx.y * 64;
    const int n0 = blockIdx.x * 16;

    // A fragment rows for the 4 M-tiles of this wave
    const __bf16* ar0 = A + (size_t)(m0 + l16) * Kpad;
    const __bf16* ar1 = ar0 + (size_t)16 * Kpad;
    const __bf16* ar2 = ar0 + (size_t)32 * Kpad;
    const __bf16* ar3 = ar0 + (size_t)48 * Kpad;
    const __bf16* brow = Bt + (size_t)(n0 + l16) * Kpad;

    // 16-bit A 16x32 layout: lanes 0-15 hold K k0+0..7 / k0+16..23,
    // lanes 16-31 hold K k0+8..15 / k0+24..31 (two 16B chunks).
    auto loadA = [&](const __bf16* row, int k0) -> v16bf {
        v16bf a;
        *(v8bf*)&a       = *(const v8bf*)(row + k0 + hf * 8);
        *((v8bf*)&a + 1) = *(const v8bf*)(row + k0 + 16 + hf * 8);
        return a;
    };
    // B 32x16: same K order as A: lanes 0-15 hold K k0+0..7 / k0+16..23,
    // lanes 16-31 hold K k0+8..15 / k0+24..31 of column n0+l16 (two 16B chunks).
    auto loadB = [&](int k0) -> v16bf {
        v16bf b;
        *(v8bf*)&b       = *(const v8bf*)(brow + k0 + hf * 8);
        *((v8bf*)&b + 1) = *(const v8bf*)(brow + k0 + 16 + hf * 8);
        return b;
    };

    v8f acc0 = {}, acc1 = {}, acc2 = {}, acc3 = {};

    // prologue: fragments for k0 = 0
    v16bf a0 = loadA(ar0, 0);
    v16bf a1 = loadA(ar1, 0);
    v16bf a2 = loadA(ar2, 0);
    v16bf a3 = loadA(ar3, 0);
    v16bf bf = loadB(0);

    for (int k0 = 32; k0 < Kpad; k0 += 32) {
        // issue next-step loads before this step's WMMA group
        v16bf na0 = loadA(ar0, k0);
        v16bf na1 = loadA(ar1, k0);
        v16bf na2 = loadA(ar2, k0);
        v16bf na3 = loadA(ar3, k0);
        v16bf nbf = loadB(k0);
        if (k0 + 32 < Kpad) {
            __builtin_prefetch(ar0 + k0 + 32, 0, 3);
            __builtin_prefetch(brow + k0 + 32, 0, 3);
        }
        // 4 independent WMMAs sharing the B fragment
        acc0 = WMMA_BF16(acc0, a0, bf);
        acc1 = WMMA_BF16(acc1, a1, bf);
        acc2 = WMMA_BF16(acc2, a2, bf);
        acc3 = WMMA_BF16(acc3, a3, bf);
        a0 = na0; a1 = na1; a2 = na2; a3 = na3; bf = nbf;
    }
    acc0 = WMMA_BF16(acc0, a0, bf);
    acc1 = WMMA_BF16(acc1, a1, bf);
    acc2 = WMMA_BF16(acc2, a2, bf);
    acc3 = WMMA_BF16(acc3, a3, bf);

    // C/D layout: lane -> N = n0 + (lane&15); VGPR r -> M = tile + r + 8*(lane>>4)
    const int n = n0 + l16;
    const float cn = colsq[n];
#pragma unroll
    for (int r = 0; r < 8; ++r) {
        int mr = m0 + r + 8 * hf;
        Y[(size_t)(mr)      * ldY + n] = acc0[r] + rowsq[mr]      + cn;
        Y[(size_t)(mr + 16) * ldY + n] = acc1[r] + rowsq[mr + 16] + cn;
        Y[(size_t)(mr + 32) * ldY + n] = acc2[r] + rowsq[mr + 32] + cn;
        Y[(size_t)(mr + 48) * ldY + n] = acc3[r] + rowsq[mr + 48] + cn;
    }
}

// ---------------------------------------------------------------------------
// Per-column BatchNorm (training stats: batch mean, biased var, eps=1e-5)
// + activation. One wave per column. act==0: ReLU -> bf16; act==1: sigmoid -> f32.
// ---------------------------------------------------------------------------
__global__ __launch_bounds__(32)
void k_bn_act(const float* __restrict__ Y, int ldY, int Bn, int N,
              const float* __restrict__ gamma, const float* __restrict__ beta,
              __bf16* __restrict__ Hout, int ldOut,
              float* __restrict__ outF, int ldF, int act) {
    int j = blockIdx.x;
    if (j >= N) return;
    int lane = threadIdx.x;

    float s = 0.0f, sq = 0.0f;
    for (int b = lane; b < Bn; b += 32) {
        float v = Y[(size_t)b * ldY + j];
        s += v;
        sq += v * v;
    }
#pragma unroll
    for (int off = 16; off > 0; off >>= 1) {
        s  += __shfl_xor(s, off, 32);
        sq += __shfl_xor(sq, off, 32);
    }
    float invB = 1.0f / (float)Bn;
    float mean = s * invB;
    float var  = sq * invB - mean * mean;
    float inv  = rsqrtf(var + 1e-5f);
    float g = gamma[j], bt = beta[j];

    for (int b = lane; b < Bn; b += 32) {
        float v = (Y[(size_t)b * ldY + j] - mean) * inv * g + bt;
        if (act == 0) {
            v = v > 0.0f ? v : 0.0f;
            Hout[(size_t)b * ldOut + j] = (__bf16)v;
        } else {
            outF[(size_t)b * ldF + j] = 1.0f / (1.0f + expf(-v));
        }
    }
}

// ---------------------------------------------------------------------------
extern "C" void kernel_launch(void* const* d_in, const int* in_sizes, int n_in,
                              void* d_out, int out_size, void* d_ws, size_t ws_size,
                              hipStream_t stream) {
    const float* X  = (const float*)d_in[0];
    const float* W1 = (const float*)d_in[1];
    const float* g1 = (const float*)d_in[2];
    const float* b1 = (const float*)d_in[3];
    const float* W2 = (const float*)d_in[4];
    const float* g2 = (const float*)d_in[5];
    const float* b2 = (const float*)d_in[6];
    const float* W3 = (const float*)d_in[7];
    const float* g3 = (const float*)d_in[8];
    const float* b3 = (const float*)d_in[9];
    const float* W4 = (const float*)d_in[10];
    const float* g4 = (const float*)d_in[11];
    const float* b4 = (const float*)d_in[12];
    float* out = (float*)d_out;

    constexpr int B    = 1024;
    constexpr int NIN  = 784, PIN  = 800;  // pad K/N to multiples of 32/16
    constexpr int N2   = 392, P2   = 416;
    constexpr int NB   = 8,   PB   = 32;
    constexpr int NOUT = 784, POUT = 800;

    char* base = (char*)d_ws;
    size_t off = 0;
    auto alloc = [&](size_t bytes) -> char* {
        char* p = base + off;
        off += (bytes + 255) & ~(size_t)255;
        return p;
    };
    __bf16* A0  = (__bf16*)alloc((size_t)B * PIN * 2);
    __bf16* H1  = (__bf16*)alloc((size_t)B * P2 * 2);   // H1,H2,H3 contiguous
    __bf16* H2  = (__bf16*)alloc((size_t)B * PB * 2);
    __bf16* H3  = (__bf16*)alloc((size_t)B * P2 * 2);
    __bf16* W1t = (__bf16*)alloc((size_t)P2 * PIN * 2);
    __bf16* W2t = (__bf16*)alloc((size_t)PB * P2 * 2);
    __bf16* W3t = (__bf16*)alloc((size_t)P2 * PB * 2);
    __bf16* W4t = (__bf16*)alloc((size_t)POUT * P2 * 2);
    float*  Y     = (float*)alloc((size_t)B * POUT * 4);
    float*  rowsq = (float*)alloc((size_t)B * 4);
    float*  colsq = (float*)alloc((size_t)POUT * 4);

    // --- stage inputs as padded bf16 ---
    k_pack_act<<<(B * PIN + 255) / 256, 256, 0, stream>>>(X, A0, B, NIN, PIN);
    k_pack_wt<<<(P2 * PIN + 255) / 256, 256, 0, stream>>>(W1, W1t, NIN, N2, PIN, P2);
    k_pack_wt<<<(PB * P2 + 255) / 256, 256, 0, stream>>>(W2, W2t, N2, NB, P2, PB);
    k_pack_wt<<<(P2 * PB + 255) / 256, 256, 0, stream>>>(W3, W3t, NB, N2, PB, P2);
    k_pack_wt<<<(POUT * P2 + 255) / 256, 256, 0, stream>>>(W4, W4t, N2, NOUT, P2, POUT);
    // zero H1|H2|H3 (contiguous) so padded columns stay zero for row norms / K pads
    {
        int nwords = (B * (P2 + PB + P2) * 2) / 4;
        k_zero32<<<(nwords + 255) / 256, 256, 0, stream>>>((uint32_t*)H1, nwords);
    }

    // --- layer 1: [1024,784] x [784,392] ---
    k_rowsq<<<(B + 127) / 128, 128, 0, stream>>>(A0, PIN, rowsq, B);
    k_colsq<<<(P2 + 127) / 128, 128, 0, stream>>>(W1, NIN, N2, P2, colsq);
    k_gemm_wmma<<<dim3(P2 / 16, B / 64), 32, 0, stream>>>(A0, W1t, rowsq, colsq, Y, PIN, P2);
    k_bn_act<<<N2, 32, 0, stream>>>(Y, P2, B, N2, g1, b1, H1, P2, nullptr, 0, 0);

    // --- layer 2: [1024,392] x [392,8] ---
    k_rowsq<<<(B + 127) / 128, 128, 0, stream>>>(H1, P2, rowsq, B);
    k_colsq<<<(PB + 127) / 128, 128, 0, stream>>>(W2, N2, NB, PB, colsq);
    k_gemm_wmma<<<dim3(PB / 16, B / 64), 32, 0, stream>>>(H1, W2t, rowsq, colsq, Y, P2, PB);
    k_bn_act<<<NB, 32, 0, stream>>>(Y, PB, B, NB, g2, b2, H2, PB, nullptr, 0, 0);

    // --- layer 3: [1024,8] x [8,392] ---
    k_rowsq<<<(B + 127) / 128, 128, 0, stream>>>(H2, PB, rowsq, B);
    k_colsq<<<(P2 + 127) / 128, 128, 0, stream>>>(W3, NB, N2, P2, colsq);
    k_gemm_wmma<<<dim3(P2 / 16, B / 64), 32, 0, stream>>>(H2, W3t, rowsq, colsq, Y, PB, P2);
    k_bn_act<<<N2, 32, 0, stream>>>(Y, P2, B, N2, g3, b3, H3, P2, nullptr, 0, 0);

    // --- layer 4: [1024,392] x [392,784], sigmoid -> fp32 d_out ---
    k_rowsq<<<(B + 127) / 128, 128, 0, stream>>>(H3, P2, rowsq, B);
    k_colsq<<<(POUT + 127) / 128, 128, 0, stream>>>(W4, N2, NOUT, POUT, colsq);
    k_gemm_wmma<<<dim3(POUT / 16, B / 64), 32, 0, stream>>>(H3, W4t, rowsq, colsq, Y, P2, POUT);
    k_bn_act<<<NOUT, 32, 0, stream>>>(Y, POUT, B, NOUT, g4, b4, nullptr, 0, out, NOUT, 1);

    (void)in_sizes; (void)n_in; (void)out_size; (void)ws_size;
}
